// TSFF_49538152792909
// MI455X (gfx1250) — matrix operand order repaired and checked by reading, not compiled, not run on hardware
//
#include <hip/hip_runtime.h>
#include <hip/hip_bf16.h>
#include <math.h>

// ---------------------------------------------------------------------------
// TSFF forward for MI455X (gfx1250, wave32).
// B=256, C=24, T=128, D=8.  All fp32 I/O; f16 only inside WMMA fragments with
// fp32 accumulation (v_wmma_f32_16x16x32_f16).
// ---------------------------------------------------------------------------

typedef __attribute__((ext_vector_type(16))) _Float16 v16h;
typedef __attribute__((ext_vector_type(8)))  float    v8f;

#define EPSBN 1e-5f

__device__ __forceinline__ float eluf(float z) {
    return (z > 0.f) ? z : (__expf(z) - 1.f);
}

// ===========================================================================
// Kernel A: time attention on x (B,1,24,128) -> h1 (B,24,128).
// q/k/v are affine in the single input channel, so:
//   logit[t1,t2] = ( a*G[t1,t2] + u*r[t1] + w*r[t2] + 24*z ) / sqrt(192)
// with G = X^T X (WMMA, K=24 padded to 32), r = column sums.
// ctx[t1,s] = sum_t2 att[t1,t2]*x[s,t2]  and rowsum via a ones-column, both
// from one WMMA pass (M=128, N=24+1, K=128).
// out[s,t] = gate[s,t]*(p*ctx[t,s] + q_*rowsum[t] + bo) + x[s,t].
// One block (128 threads = 4 waves) per batch.  Two independent WMMAs are
// kept in flight per wave to cover the WMMA->VALU hazard window; fragment
// gathers are contiguous (xhT for K-major reads) so they merge into wide
// ds_load ops.
// ===========================================================================
__global__ __launch_bounds__(128) void attn_t_kernel(
    const float* __restrict__ x, float* __restrict__ out,
    const float* __restrict__ wq, const float* __restrict__ bq,
    const float* __restrict__ wk, const float* __restrict__ bk,
    const float* __restrict__ wv, const float* __restrict__ bv,
    const float* __restrict__ wo, const float* __restrict__ bo,
    const float* __restrict__ gate)
{
    __shared__ _Float16 xh[24 * 128];     // x row-major [s][t] (ctx-phase B)
    __shared__ _Float16 xhT[128 * 32];    // x transposed [t][s], cols 24..31 = 0
    __shared__ float    r[128];           // column sums
    __shared__ _Float16 att[128 * 128];   // logits then probabilities (f16)
    __shared__ float    ctx[128 * 26];    // ctx[t1][0..23]=sum, [24]=rowsum

    const int b   = blockIdx.x;
    const int tid = threadIdx.x;
    const float* xb = x + b * 3072;

    for (int i = tid; i < 3072; i += 128) {
        const float v = xb[i];
        const int s = i >> 7, t = i & 127;
        xh[i] = (_Float16)v;
        xhT[t * 32 + s] = (_Float16)v;
    }
    for (int i = tid; i < 1024; i += 128) {        // zero K-pad cols 24..31
        const int t = i >> 3, s = 24 + (i & 7);
        xhT[t * 32 + s] = (_Float16)0.f;
    }
    __syncthreads();

    { // column sums r[t] = sum_s x[s,t]
        float s = 0.f;
        for (int ss = 0; ss < 24; ++ss) s += (float)xh[ss * 128 + tid];
        r[tid] = s;
    }
    // scalar param contractions (redundant per thread, 8 elements)
    float a_ = 0.f, u_ = 0.f, w_ = 0.f, z_ = 0.f, p_ = 0.f, q_ = 0.f;
    for (int d = 0; d < 8; ++d) {
        a_ += wq[d] * wk[d];  u_ += wq[d] * bk[d];
        w_ += bq[d] * wk[d];  z_ += bq[d] * bk[d];
        p_ += wo[d] * wv[d];  q_ += wo[d] * bv[d];
    }
    const float bo0   = bo[0];
    const float invsc = 1.f / sqrtf(192.f);
    __syncthreads();

    const int wave = tid >> 5;
    const int lane = tid & 31;
    const int lm   = lane & 15;
    const int hi   = (lane >> 4) & 1;

    // ---- G = X^T X, 64 tiles of 16x16; each wave runs two tiles at a time.
    for (int q = 0; q < 16; q += 2) {
        const int tA = wave + 4 * q;
        const int tB = wave + 4 * (q + 1);
        const int tmA = (tA >> 3) * 16, tnA = (tA & 7) * 16;
        const int tmB = (tB >> 3) * 16, tnB = (tB & 7) * 16;
        v16h a0, b0, a1, b1;
#pragma unroll
        for (int j = 0; j < 8; ++j) {
            // A layout (16-bit 16x32): lanes0-15 K=0..7 & 16..23; lanes16-31 +8
            const int ka = ((j < 4) ? (2 * j) : (16 + 2 * (j - 4))) + (hi ? 8 : 0);
            // B layout (16-bit 32x16): lanes0-15 K=0..7 & 16..23; lanes16-31 +8
            const int kb = ((j < 4) ? (2 * j) : (16 + 2 * (j - 4))) + (hi ? 8 : 0);
            a0[2 * j]     = xhT[(tmA + lm) * 32 + ka];     // A[t1,k]=x[k,t1]
            a0[2 * j + 1] = xhT[(tmA + lm) * 32 + ka + 1];
            b0[2 * j]     = xhT[(tnA + lm) * 32 + kb];     // B[k,t2]=x[k,t2]
            b0[2 * j + 1] = xhT[(tnA + lm) * 32 + kb + 1];
            a1[2 * j]     = xhT[(tmB + lm) * 32 + ka];
            a1[2 * j + 1] = xhT[(tmB + lm) * 32 + ka + 1];
            b1[2 * j]     = xhT[(tnB + lm) * 32 + kb];
            b1[2 * j + 1] = xhT[(tnB + lm) * 32 + kb + 1];
        }
        v8f c0 = {}, c1 = {};
        c0 = __builtin_amdgcn_wmma_f32_16x16x32_f16(false, a0, false, b0,
                                                    (short)0, c0, false, false);
        c1 = __builtin_amdgcn_wmma_f32_16x16x32_f16(false, a1, false, b1,
                                                    (short)0, c1, false, false);
#pragma unroll
        for (int rr = 0; rr < 8; ++rr) {
            // C/D layout: vgpr = M (lanes16-31: M+8), lane = N
            const int mrA = tmA + rr + (hi ? 8 : 0), ncA = tnA + lm;
            const int mrB = tmB + rr + (hi ? 8 : 0), ncB = tnB + lm;
            const float lA = (a_ * c0[rr] + u_ * r[mrA] + w_ * r[ncA] + 24.f * z_) * invsc;
            const float lB = (a_ * c1[rr] + u_ * r[mrB] + w_ * r[ncB] + 24.f * z_) * invsc;
            att[mrA * 128 + ncA] = (_Float16)lA;
            att[mrB * 128 + ncB] = (_Float16)lB;
        }
    }
    __syncthreads();

    // ---- softmax1 per row (thread t owns row t), staggered to spread banks
    {
        const int row = tid;
        float mx = -1e30f;
        for (int j = 0; j < 128; ++j) {
            const int jj = (j + row) & 127;
            mx = fmaxf(mx, (float)att[row * 128 + jj]);
        }
        float s = 0.f;
        for (int j = 0; j < 128; ++j) {
            const int jj = (j + row) & 127;
            s += __expf((float)att[row * 128 + jj] - mx);
        }
        const float inv = 1.f / (1.f + s);
        for (int j = 0; j < 128; ++j) {
            const int jj = (j + row) & 127;
            att[row * 128 + jj] =
                (_Float16)(__expf((float)att[row * 128 + jj] - mx) * inv);
        }
    }
    __syncthreads();

    // ---- ctx = att * X^T (+ ones column 24 -> rowsum).
    // 8 M-tiles x 2 N-tiles; each wave: 2 M-tiles, both N-tiles concurrently
    // (shared A fragment, two accumulators -> back-to-back WMMAs).
    for (int mi = 0; mi < 2; ++mi) {
        const int tm = (wave + 4 * mi) * 16;
        const int m  = tm + lm;
        const int n0 = lm;        // N-tile 0: s = 0..15
        const int n1 = 16 + lm;   // N-tile 1: s = 16..23, 24 = ones, rest 0
        v8f c0 = {}, c1 = {};
        for (int ks = 0; ks < 4; ++ks) {
            v16h av, bv0, bv1;
#pragma unroll
            for (int j = 0; j < 8; ++j) {
                const int ka = ks * 32 + ((j < 4) ? (2 * j) : (16 + 2 * (j - 4))) + (hi ? 8 : 0);
                av[2 * j]     = att[m * 128 + ka];
                av[2 * j + 1] = att[m * 128 + ka + 1];
                const int kb = ks * 32 + ((j < 4) ? (2 * j) : (16 + 2 * (j - 4))) + (hi ? 8 : 0);
                bv0[2 * j]     = xh[n0 * 128 + kb];      // B[k,s]=x[s,k]
                bv0[2 * j + 1] = xh[n0 * 128 + kb + 1];
                _Float16 e0, e1;
                if (n1 < 24)       { e0 = xh[n1 * 128 + kb]; e1 = xh[n1 * 128 + kb + 1]; }
                else if (n1 == 24) { e0 = (_Float16)1.f;     e1 = (_Float16)1.f; }
                else               { e0 = (_Float16)0.f;     e1 = (_Float16)0.f; }
                bv1[2 * j] = e0; bv1[2 * j + 1] = e1;
            }
            c0 = __builtin_amdgcn_wmma_f32_16x16x32_f16(false, av, false, bv0,
                                                        (short)0, c0, false, false);
            c1 = __builtin_amdgcn_wmma_f32_16x16x32_f16(false, av, false, bv1,
                                                        (short)0, c1, false, false);
        }
#pragma unroll
        for (int rr = 0; rr < 8; ++rr) {
            const int mr = tm + rr + (hi ? 8 : 0);
            ctx[mr * 26 + lm] = c0[rr];
            if (n1 < 25) ctx[mr * 26 + n1] = c1[rr];
        }
    }
    __syncthreads();

    for (int i = tid; i < 3072; i += 128) {
        const int s = i >> 7, t = i & 127;
        const float y = p_ * ctx[t * 26 + s] + q_ * ctx[t * 26 + 24] + bo0;
        out[b * 3072 + i] = gate[i] * y + xb[i];
    }
}

// ===========================================================================
// Kernel B: 125-tap temporal conv + BN: (B,24,128) -> (B,8,24,128).
// One block per (b,s); 128 threads over t.
// ===========================================================================
__global__ __launch_bounds__(128) void conv125_bn_kernel(
    const float* __restrict__ in, float* __restrict__ out,
    const float* __restrict__ w,   // (8,125)
    const float* __restrict__ g, const float* __restrict__ bt,
    const float* __restrict__ mu, const float* __restrict__ var)
{
    __shared__ float xp[252];
    __shared__ float wl[1000];
    const int blk = blockIdx.x;
    const int b = blk / 24, s = blk % 24;
    const int t = threadIdx.x;
    const float* row = in + (b * 24 + s) * 128;
    for (int i = t; i < 252; i += 128) {
        const int xi = i - 62;
        xp[i] = (xi >= 0 && xi < 128) ? row[xi] : 0.f;
    }
    for (int i = t; i < 1000; i += 128) wl[i] = w[i];
    __syncthreads();
    for (int d = 0; d < 8; ++d) {
        float acc = 0.f;
        const float* wd = &wl[d * 125];
        for (int k = 0; k < 125; ++k) acc = fmaf(wd[k], xp[t + k], acc);
        const float sc = g[d] * rsqrtf(var[d] + EPSBN);
        out[((b * 8 + d) * 24 + s) * 128 + t] = (acc - mu[d]) * sc + bt[d];
    }
}

// ===========================================================================
// Kernel C: space attention over s (generic ci=co in {1,8}, inner dim 8).
// One block (256 threads) per batch.  q/k computed in two e-chunks (LDS<=64KB),
// buffer reused for v.  May run in-place (hout==hin).
// ===========================================================================
__global__ __launch_bounds__(256) void attn_s_kernel(
    const float* __restrict__ hin, float* __restrict__ hout, int ci,
    const float* __restrict__ wq, const float* __restrict__ bq,
    const float* __restrict__ wk, const float* __restrict__ bk,
    const float* __restrict__ wv, const float* __restrict__ bv,
    const float* __restrict__ wo, const float* __restrict__ bo,
    const float* __restrict__ gate, float scale)
{
    __shared__ _Float16 sb[24576];  // q-chunk[0..12287] | k-chunk[12288..]; later v[8][3072]
    __shared__ float attf[600];     // att[24][25]
    __shared__ float pq[64], pk[64], pv[64], po[64];
    __shared__ float pbq[8], pbk[8], pbv[8], pbo[8];

    const int b = blockIdx.x, tid = threadIdx.x;
    const float* hb = hin + (size_t)b * ci * 3072;

    if (tid < 8 * ci) { pq[tid] = wq[tid]; pk[tid] = wk[tid]; pv[tid] = wv[tid]; }
    if (tid < ci * 8) po[tid] = wo[tid];
    if (tid < 8) { pbq[tid] = bq[tid]; pbk[tid] = bk[tid]; pbv[tid] = bv[tid]; }
    if (tid < ci) pbo[tid] = bo[tid];
    for (int i = tid; i < 600; i += 256) attf[i] = 0.f;
    __syncthreads();

    for (int chunk = 0; chunk < 2; ++chunk) {
        for (int i = tid; i < 3072; i += 256) {
            float hv[8];
            for (int ii = 0; ii < ci; ++ii) hv[ii] = hb[ii * 3072 + i];
            for (int eo = 0; eo < 4; ++eo) {
                const int e = chunk * 4 + eo;
                float qv = pbq[e], kv = pbk[e];
                for (int ii = 0; ii < ci; ++ii) {
                    qv = fmaf(pq[e * ci + ii], hv[ii], qv);
                    kv = fmaf(pk[e * ci + ii], hv[ii], kv);
                }
                sb[eo * 3072 + i]         = (_Float16)qv;
                sb[12288 + eo * 3072 + i] = (_Float16)kv;
            }
        }
        __syncthreads();
        for (int p = tid; p < 576; p += 256) {
            const int s1 = p / 24, s2 = p % 24;
            float acc = 0.f;
            for (int eo = 0; eo < 4; ++eo) {
                const _Float16* qr = &sb[eo * 3072 + s1 * 128];
                const _Float16* kr = &sb[12288 + eo * 3072 + s2 * 128];
                for (int t = 0; t < 128; ++t)
                    acc = fmaf((float)qr[t], (float)kr[t], acc);
            }
            attf[s1 * 25 + s2] += acc;
        }
        __syncthreads();
    }
    if (tid < 24) { // softmax1 row
        float mx = -1e30f;
        for (int j = 0; j < 24; ++j) mx = fmaxf(mx, attf[tid * 25 + j] / scale);
        float ex[24], s = 0.f;
        for (int j = 0; j < 24; ++j) { ex[j] = __expf(attf[tid * 25 + j] / scale - mx); s += ex[j]; }
        const float inv = 1.f / (1.f + s);
        for (int j = 0; j < 24; ++j) attf[tid * 25 + j] = ex[j] * inv;
    }
    __syncthreads();
    for (int i = tid; i < 3072; i += 256) { // v (bias folded in)
        float hv[8];
        for (int ii = 0; ii < ci; ++ii) hv[ii] = hb[ii * 3072 + i];
        for (int e = 0; e < 8; ++e) {
            float vv = pbv[e];
            for (int ii = 0; ii < ci; ++ii) vv = fmaf(pv[e * ci + ii], hv[ii], vv);
            sb[e * 3072 + i] = (_Float16)vv;
        }
    }
    __syncthreads();
    for (int i = tid; i < 3072; i += 256) {
        const int s1 = i >> 7, t = i & 127;
        float y[8];
        for (int e = 0; e < 8; ++e) {
            float acc = 0.f;
            const float* ar = &attf[s1 * 25];
            const _Float16* ve = &sb[e * 3072 + t];
            for (int s2 = 0; s2 < 24; ++s2) acc = fmaf(ar[s2], (float)ve[s2 * 128], acc);
            y[e] = acc;
        }
        for (int o = 0; o < ci; ++o) {
            float val = pbo[o];
            for (int e = 0; e < 8; ++e) val = fmaf(po[o * 8 + e], y[e], val);
            const int gi = o * 3072 + i;
            hout[(size_t)b * ci * 3072 + gi] = gate[gi] * val + hb[gi];
        }
    }
}

// ===========================================================================
// Kernel D: einsum('bist,ois->bot') + (BN|bias) + ELU + pool4:
// (B,8,24,128) -> (B,16,32).  One block per batch, 128 threads over t.
// ===========================================================================
__global__ __launch_bounds__(128) void reduce_pool_kernel(
    const float* __restrict__ hin, float* __restrict__ pooled,
    const float* __restrict__ w,   // (16,8,24)
    const float* __restrict__ g, const float* __restrict__ bt,
    const float* __restrict__ mu, const float* __restrict__ var,
    const float* __restrict__ bias, int use_bn)
{
    __shared__ float wl[3072];
    __shared__ float ebuf[16][128];
    const int b = blockIdx.x, t = threadIdx.x;
    for (int i = t; i < 3072; i += 128) wl[i] = w[i];
    __syncthreads();
    const float* hb = hin + (size_t)b * 24576;
    float acc[16];
    for (int o = 0; o < 16; ++o) acc[o] = 0.f;
    for (int is = 0; is < 192; ++is) {
        const float hv = hb[is * 128 + t];
        for (int o = 0; o < 16; ++o) acc[o] = fmaf(wl[o * 192 + is], hv, acc[o]);
    }
    for (int o = 0; o < 16; ++o) {
        float z;
        if (use_bn) {
            const float sc = g[o] * rsqrtf(var[o] + EPSBN);
            z = (acc[o] - mu[o]) * sc + bt[o];
        } else z = acc[o] + bias[o];
        ebuf[o][t] = eluf(z);
    }
    __syncthreads();
    for (int i = t; i < 512; i += 128) {
        const int o = i >> 5, tp = i & 31;
        pooled[b * 512 + i] = 0.25f * (ebuf[o][tp * 4] + ebuf[o][tp * 4 + 1] +
                                       ebuf[o][tp * 4 + 2] + ebuf[o][tp * 4 + 3]);
    }
}

// ===========================================================================
// Kernel E: conv15 + 1x1 + BN + ELU + pool8: (B,16,32) -> (B,64).
// ===========================================================================
__global__ __launch_bounds__(128) void conv15_head_kernel(
    const float* __restrict__ pooled, float* __restrict__ outvec,
    const float* __restrict__ w1,  // (16,16,15)
    const float* __restrict__ w2,  // (16,16)
    const float* __restrict__ g, const float* __restrict__ bt,
    const float* __restrict__ mu, const float* __restrict__ var)
{
    __shared__ float xp[16][46];
    __shared__ float c1[16][32];
    __shared__ float e2[16][32];
    __shared__ float wl1[3840];
    __shared__ float wl2[256];
    const int b = blockIdx.x, tid = threadIdx.x;
    for (int i = tid; i < 3840; i += 128) wl1[i] = w1[i];
    for (int i = tid; i < 256; i += 128) wl2[i] = w2[i];
    for (int i = tid; i < 16 * 46; i += 128) {
        const int o = i / 46, t = i % 46, xi = t - 7;
        xp[o][t] = (xi >= 0 && xi < 32) ? pooled[b * 512 + o * 32 + xi] : 0.f;
    }
    __syncthreads();
    for (int i = tid; i < 512; i += 128) {
        const int o = i >> 5, t = i & 31;
        float acc = 0.f;
        for (int ii = 0; ii < 16; ++ii) {
            const float* wr = &wl1[(o * 16 + ii) * 15];
            const float* xr = &xp[ii][t];
            for (int k = 0; k < 15; ++k) acc = fmaf(wr[k], xr[k], acc);
        }
        c1[o][t] = acc;
    }
    __syncthreads();
    for (int i = tid; i < 512; i += 128) {
        const int o = i >> 5, t = i & 31;
        float acc = 0.f;
        for (int ii = 0; ii < 16; ++ii) acc = fmaf(wl2[o * 16 + ii], c1[ii][t], acc);
        const float sc = g[o] * rsqrtf(var[o] + EPSBN);
        e2[o][t] = eluf((acc - mu[o]) * sc + bt[o]);
    }
    __syncthreads();
    if (tid < 64) {
        const int o = tid >> 2, tp = tid & 3;
        float s = 0.f;
        for (int k = 0; k < 8; ++k) s += e2[o][tp * 8 + k];
        outvec[b * 64 + tid] = 0.125f * s;
    }
}

// ===========================================================================
// Kernel F: frequency front — the 75 MB streamer.
// f0 = elu(W3b * BN8(W3a * xf + b3a) + b3b): (B,24,24,128) -> (B,24,128).
// One block per (b,s); 128 threads over t; 24 coalesced loads/thread.
// ===========================================================================
__global__ __launch_bounds__(128) void freq_front_kernel(
    const float* __restrict__ xf, float* __restrict__ f0,
    const float* __restrict__ w3a, const float* __restrict__ b3a,
    const float* __restrict__ w3b, const float* __restrict__ b3b,
    const float* __restrict__ g, const float* __restrict__ bt,
    const float* __restrict__ mu, const float* __restrict__ var)
{
    __shared__ float wl[192];
    const int blk = blockIdx.x;
    const int b = blk / 24, s = blk % 24;
    const int t = threadIdx.x;
    for (int i = t; i < 192; i += 128) wl[i] = w3a[i];
    __syncthreads();
    const float* xb = xf + (size_t)b * 73728 + s * 128;
    float acc[8] = {0.f, 0.f, 0.f, 0.f, 0.f, 0.f, 0.f, 0.f};
    for (int i = 0; i < 24; ++i) {
        const float hv = xb[(size_t)i * 3072 + t];
        for (int e = 0; e < 8; ++e) acc[e] = fmaf(wl[e * 24 + i], hv, acc[e]);
    }
    float o = b3b[0];
    for (int e = 0; e < 8; ++e) {
        const float sc = g[e] * rsqrtf(var[e] + EPSBN);
        const float z = (acc[e] + b3a[e] - mu[e]) * sc + bt[e];
        o = fmaf(w3b[e], z, o);
    }
    f0[(b * 24 + s) * 128 + t] = eluf(o);
}

// ===========================================================================
// Kernel G: fuse + MLP head + classifier: (B,64)x2 -> (B,3).
// ===========================================================================
__global__ __launch_bounds__(128) void head_kernel(
    const float* __restrict__ st, const float* __restrict__ fq,
    const float* __restrict__ wi,
    const float* __restrict__ w1, const float* __restrict__ b1,
    const float* __restrict__ w2, const float* __restrict__ b2,
    const float* __restrict__ wc, const float* __restrict__ bc,
    float* __restrict__ out)
{
    __shared__ float fz[64], z1[128], z2[64];
    const int b = blockIdx.x, tid = threadIdx.x;
    const float wiv = wi[0];
    if (tid < 64) fz[tid] = wiv * st[b * 64 + tid] + (1.f - wiv) * fq[b * 64 + tid];
    __syncthreads();
    {
        float acc = b1[tid];
        for (int i = 0; i < 64; ++i) acc = fmaf(fz[i], w1[i * 128 + tid], acc);
        z1[tid] = eluf(acc);
    }
    __syncthreads();
    if (tid < 64) {
        float acc = b2[tid];
        for (int i = 0; i < 128; ++i) acc = fmaf(z1[i], w2[i * 64 + tid], acc);
        z2[tid] = acc;
    }
    __syncthreads();
    if (tid < 3) {
        float acc = bc[tid];
        for (int i = 0; i < 64; ++i) acc = fmaf(z2[i], wc[i * 3 + tid], acc);
        out[b * 3 + tid] = acc;
    }
}

// ===========================================================================
// Launch.  Input order = setup_inputs() insertion order: x, x_frequency, then
// params leaves depth-first in dict insertion order.
// ===========================================================================
#define PIN(i) ((const float*)d_in[(i)])

extern "C" void kernel_launch(void* const* d_in, const int* in_sizes, int n_in,
                              void* d_out, int out_size, void* d_ws, size_t ws_size,
                              hipStream_t stream) {
    (void)in_sizes; (void)n_in; (void)out_size; (void)ws_size;
    const float* x  = PIN(0);   // (256,1,24,128)
    const float* xf = PIN(1);   // (256,24,24,128)
    // 2..10:  ta.{wq,bq,wk,bk,wv,bv,wo,bo,gate}
    // 11..15: b1.{w,g,b,m,v}
    // 16..24: sa.{wq,bq,wk,bk,wv,bv,wo,bo,gate}
    // 25..29: b2.{w,g,b,m,v}
    // 30..35: b3.{w1,w2,g,b,m,v}
    // 36..52: fa.{w3a,b3a,w3b,b3b,g,b,m,v,wq,bq,wk,bk,wv,bv,wo,bo,gate}
    // 53..65: fe.{w1,bn1.g,bn1.b,bn1.m,bn1.v,w2,b2,w3,w4,bn2.g,bn2.b,bn2.m,bn2.v}
    // 66: wi; 67..70: ff.{w1,b1,w2,b2}; 71..72: cls.{w,b}

    char* ws = (char*)d_ws;
    float* h1     = (float*)(ws + 0);          //  3,145,728 B  (B,24,128)
    float* h2     = (float*)(ws + 3145728);    // 25,165,824 B  (B,8,24,128)
    float* pooled = (float*)(ws + 28311552);   //    524,288 B  (B,16,32)
    float* st     = (float*)(ws + 28835840);   //     65,536 B  (B,64)
    float* f0     = (float*)(ws + 28901376);   //  3,145,728 B
    float* f1     = (float*)(ws + 32047104);   //  3,145,728 B
    float* fq     = (float*)(ws + 35192832);   //     65,536 B

    // ---- time-space branch ----
    attn_t_kernel<<<256, 128, 0, stream>>>(x, h1,
        PIN(2), PIN(3), PIN(4), PIN(5), PIN(6), PIN(7), PIN(8), PIN(9), PIN(10));
    conv125_bn_kernel<<<6144, 128, 0, stream>>>(h1, h2,
        PIN(11), PIN(12), PIN(13), PIN(14), PIN(15));
    attn_s_kernel<<<256, 256, 0, stream>>>(h2, h2, 8,
        PIN(16), PIN(17), PIN(18), PIN(19), PIN(20), PIN(21), PIN(22), PIN(23),
        PIN(24), 32.0f);
    reduce_pool_kernel<<<256, 128, 0, stream>>>(h2, pooled,
        PIN(25), PIN(26), PIN(27), PIN(28), PIN(29), PIN(26), 1);
    conv15_head_kernel<<<256, 128, 0, stream>>>(pooled, st,
        PIN(30), PIN(31), PIN(32), PIN(33), PIN(34), PIN(35));

    // ---- frequency branch ----
    freq_front_kernel<<<6144, 128, 0, stream>>>(xf, f0,
        PIN(36), PIN(37), PIN(38), PIN(39), PIN(40), PIN(41), PIN(42), PIN(43));
    attn_s_kernel<<<256, 256, 0, stream>>>(f0, f1, 1,
        PIN(44), PIN(45), PIN(46), PIN(47), PIN(48), PIN(49), PIN(50), PIN(51),
        PIN(52), 32.0f);
    conv125_bn_kernel<<<6144, 128, 0, stream>>>(f1, h2,
        PIN(53), PIN(54), PIN(55), PIN(56), PIN(57));
    reduce_pool_kernel<<<256, 128, 0, stream>>>(h2, pooled,
        PIN(58), PIN(54), PIN(55), PIN(56), PIN(57), PIN(59), 0);
    conv15_head_kernel<<<256, 128, 0, stream>>>(pooled, fq,
        PIN(60), PIN(61), PIN(62), PIN(63), PIN(64), PIN(65));

    // ---- fuse + head ----
    head_kernel<<<256, 128, 0, stream>>>(st, fq, PIN(66),
        PIN(67), PIN(68), PIN(69), PIN(70), PIN(71), PIN(72), (float*)d_out);
}
